// SimpleCausalLM_8211977470189
// MI455X (gfx1250) — compile-verified, matrix operand order checked by reading, not run on hardware
//
#include <hip/hip_runtime.h>
#include <math.h>

typedef __bf16 bf16;
typedef __attribute__((ext_vector_type(16))) bf16 v16bf;
typedef __attribute__((ext_vector_type(8)))  bf16 v8bf;
typedef __attribute__((ext_vector_type(8)))  float v8f;
typedef __attribute__((ext_vector_type(4)))  int   v4i;

#define S_LEN 2048
#define D_DIM 2048
#define F_DIM 8192
#define V_DIM 32000
#define LN_EPS 1e-5f

// ---------------- GEMM config: C[M,N] = A[M,K] @ B[N,K]^T  (both K-contiguous)
#define BM 128
#define BN 128
#define BK 32
#define GTHREADS 256
#define LDT (BK + 8)       // LDS row stride in bf16 => 80 bytes, 16B aligned

// ---- CDNA5 async global->LDS copy (ASYNCcnt path), with sync fallback
#if defined(__has_builtin)
# if __has_builtin(__builtin_amdgcn_global_load_async_to_lds_b128)
#  define HAS_ASYNC_LDS 1
# endif
# if __has_builtin(__builtin_amdgcn_s_wait_asynccnt)
#  define HAS_WAIT_ASYNC 1
# endif
#endif

__device__ __forceinline__ void wait_async0() {
#if defined(HAS_WAIT_ASYNC)
  __builtin_amdgcn_s_wait_asynccnt(0);
#else
  asm volatile("s_wait_asynccnt 0x0" ::: "memory");
#endif
}
// async loads retire in order: <=4 outstanding guarantees all but the newest tile landed
__device__ __forceinline__ void wait_async4() {
#if defined(HAS_WAIT_ASYNC)
  __builtin_amdgcn_s_wait_asynccnt(4);
#else
  asm volatile("s_wait_asynccnt 0x4" ::: "memory");
#endif
}

#if defined(HAS_ASYNC_LDS)
__device__ __forceinline__ void async_cp16(const bf16* g, bf16* l) {
  __builtin_amdgcn_global_load_async_to_lds_b128(
      (__attribute__((address_space(1))) v4i*)g,
      (__attribute__((address_space(3))) v4i*)l,
      0, 0);
}
#endif

// Load one 16x32 bf16 fragment row per the CDNA5 16-bit A/B layout:
// lanes 0-15 hold K=0..7 (vgpr0-3) and K=16..23 (vgpr4-7); lanes 16-31 hold K=8..15 / K=24..31.
__device__ __forceinline__ v16bf frag16(const bf16* rowp, int half) {
  const v8bf lo = *(const v8bf*)(rowp + half * 8);
  const v8bf hi = *(const v8bf*)(rowp + 16 + half * 8);
  v16bf r;
#pragma unroll
  for (int i = 0; i < 8; ++i) { r[i] = lo[i]; r[i + 8] = hi[i]; }
  return r;
}

// Compile-time epilogue flags -> branch-free epilogues per GEMM variant.
// CSKIP: skip blocks entirely above the causal diagonal (scores GEMM).
// CK:    clip the K loop at m0+BM (AV GEMM; attn rows are zero past diagonal).
template <bool BIAS, bool RESID, bool RELU, bool OUTF, bool OUTB, bool CSKIP, bool CK>
__global__ __launch_bounds__(GTHREADS) void gemm_bf16_kernel(
    const bf16* __restrict__ A, const bf16* __restrict__ B,
    float* __restrict__ outF, bf16* __restrict__ outB,
    const float* __restrict__ bias, const float* __restrict__ resid,
    int N, int K, float scale)
{
  const int m0 = (int)blockIdx.y * BM;
  const int n0 = (int)blockIdx.x * BN;
  if (CSKIP && n0 >= m0 + BM) return;   // fully-masked causal block

  __shared__ bf16 As[3][BM * LDT];
  __shared__ bf16 Bs[3][BN * LDT];

  const int tid  = threadIdx.x;
  const int lane = tid & 31;           // wave32
  const int wid  = tid >> 5;           // 8 waves
  const int l    = lane & 15;
  const int half = lane >> 4;
  const int wm   = wid >> 2;           // 0..1 : 64-row strip
  const int wn   = wid & 3;            // 0..3 : 32-col strip

  // global->LDS staging: thread owns row=tid/2, 16 bf16 at col=(tid&1)*16
  const int ldRow = tid >> 1;
  const int ldCol = (tid & 1) * 16;
  const int sOff  = ldRow * LDT + ldCol;
  const bf16* Ag = A + (long)(m0 + ldRow) * K + ldCol;
  const bf16* Bg = B + (long)(n0 + ldRow) * K + ldCol;

  v8f acc[4][2];
#pragma unroll
  for (int i = 0; i < 4; ++i)
#pragma unroll
    for (int j = 0; j < 2; ++j)
      acc[i][j] = (v8f){0.f, 0.f, 0.f, 0.f, 0.f, 0.f, 0.f, 0.f};

  const int kmax = CK ? (m0 + BM) : K;
  const int nk   = kmax / BK;

#if defined(HAS_ASYNC_LDS)
  // ---- 3-stage ASYNCcnt-tracked global->LDS DMA pipeline (2 tiles in flight)
  {
    async_cp16(Ag,     &As[0][sOff]);
    async_cp16(Ag + 8, &As[0][sOff + 8]);
    async_cp16(Bg,     &Bs[0][sOff]);
    async_cp16(Bg + 8, &Bs[0][sOff + 8]);
  }
  if (nk > 1) {
    const bf16* Agn = Ag + BK;
    const bf16* Bgn = Bg + BK;
    async_cp16(Agn,     &As[1][sOff]);
    async_cp16(Agn + 8, &As[1][sOff + 8]);
    async_cp16(Bgn,     &Bs[1][sOff]);
    async_cp16(Bgn + 8, &Bs[1][sOff + 8]);
    wait_async4();          // tile 0 complete (in-order retirement)
  } else {
    wait_async0();
  }
  __syncthreads();

  int cur = 0, ib = 2;      // ib = buffer to issue into (consumed in the previous iter)
  for (int kt = 0; kt < nk; ++kt) {
    const bool more2 = (kt + 2) < nk;
    if (more2) {
      const bf16* Agn = Ag + (long)(kt + 2) * BK;
      const bf16* Bgn = Bg + (long)(kt + 2) * BK;
      async_cp16(Agn,     &As[ib][sOff]);
      async_cp16(Agn + 8, &As[ib][sOff + 8]);
      async_cp16(Bgn,     &Bs[ib][sOff]);
      async_cp16(Bgn + 8, &Bs[ib][sOff + 8]);
    }

    v16bf af[4], bfr[2];
#pragma unroll
    for (int tm = 0; tm < 4; ++tm)
      af[tm] = frag16(&As[cur][(wm * 64 + tm * 16 + l) * LDT], half);
#pragma unroll
    for (int tn = 0; tn < 2; ++tn)
      bfr[tn] = frag16(&Bs[cur][(wn * 32 + tn * 16 + l) * LDT], half);
#pragma unroll
    for (int tm = 0; tm < 4; ++tm)
#pragma unroll
      for (int tn = 0; tn < 2; ++tn)
        acc[tm][tn] = __builtin_amdgcn_wmma_f32_16x16x32_bf16(
            false, af[tm], false, bfr[tn], (short)0, acc[tm][tn], false, false);

    if (more2) wait_async4();   // tile kt+1 landed; tile kt+2 may stay in flight
    else       wait_async0();
    __syncthreads();

    const int t = cur;
    cur = (cur == 2) ? 0 : cur + 1;
    ib  = t;
  }
#else
  // ---- fallback: VGPR-staged double buffer (uses stages 0/1)
  {
    v8bf a0 = *(const v8bf*)(Ag);
    v8bf a1 = *(const v8bf*)(Ag + 8);
    v8bf b0 = *(const v8bf*)(Bg);
    v8bf b1 = *(const v8bf*)(Bg + 8);
    *(v8bf*)(&As[0][sOff])     = a0;
    *(v8bf*)(&As[0][sOff + 8]) = a1;
    *(v8bf*)(&Bs[0][sOff])     = b0;
    *(v8bf*)(&Bs[0][sOff + 8]) = b1;
  }
  __syncthreads();

  for (int kt = 0; kt < nk; ++kt) {
    const int  cur  = kt & 1;
    const int  nxt  = cur ^ 1;
    const bool more = (kt + 1) < nk;
    v8bf a0, a1, b0, b1;
    if (more) {
      const bf16* Agn = Ag + (long)(kt + 1) * BK;
      const bf16* Bgn = Bg + (long)(kt + 1) * BK;
      a0 = *(const v8bf*)(Agn);
      a1 = *(const v8bf*)(Agn + 8);
      b0 = *(const v8bf*)(Bgn);
      b1 = *(const v8bf*)(Bgn + 8);
    }

    v16bf af[4], bfr[2];
#pragma unroll
    for (int tm = 0; tm < 4; ++tm)
      af[tm] = frag16(&As[cur][(wm * 64 + tm * 16 + l) * LDT], half);
#pragma unroll
    for (int tn = 0; tn < 2; ++tn)
      bfr[tn] = frag16(&Bs[cur][(wn * 32 + tn * 16 + l) * LDT], half);
#pragma unroll
    for (int tm = 0; tm < 4; ++tm)
#pragma unroll
      for (int tn = 0; tn < 2; ++tn)
        acc[tm][tn] = __builtin_amdgcn_wmma_f32_16x16x32_bf16(
            false, af[tm], false, bfr[tn], (short)0, acc[tm][tn], false, false);

    if (more) {
      *(v8bf*)(&As[nxt][sOff])     = a0;
      *(v8bf*)(&As[nxt][sOff + 8]) = a1;
      *(v8bf*)(&Bs[nxt][sOff])     = b0;
      *(v8bf*)(&Bs[nxt][sOff + 8]) = b1;
    }
    __syncthreads();
  }
#endif

  // ---- branch-free epilogue (flags compile-time; 32-bit index math, all < 2^31)
#pragma unroll
  for (int tn = 0; tn < 2; ++tn) {
    const int col   = n0 + wn * 32 + tn * 16 + l;
    const float bcol = BIAS ? bias[col] : 0.f;
#pragma unroll
    for (int tm = 0; tm < 4; ++tm) {
      const int row0 = m0 + wm * 64 + tm * 16 + half * 8;
      int idx = row0 * N + col;
#pragma unroll
      for (int r = 0; r < 8; ++r) {
        float v = acc[tm][tn][r] * scale;
        if (BIAS)  v += bcol;
        if (RESID) v += resid[idx];
        if (RELU)  v = fmaxf(v, 0.f);
        if (OUTF)  outF[idx] = v;
        if (OUTB)  outB[idx] = (bf16)v;
        idx += N;
      }
    }
  }
}

// ---------------- elementwise / reduction kernels ----------------

__global__ __launch_bounds__(256) void embed_kernel(
    const int* __restrict__ tokens, const float* __restrict__ tok_emb,
    const float* __restrict__ pos_emb, float* __restrict__ x)
{
  const int s = blockIdx.x;
  const long t = tokens[s];
#pragma unroll
  for (int i = 0; i < D_DIM / 256; ++i) {
    const int d = threadIdx.x + i * 256;
    x[(long)s * D_DIM + d] = tok_emb[t * D_DIM + d] + pos_emb[(long)s * D_DIM + d];
  }
}

__global__ __launch_bounds__(256) void cvt_bf16_kernel(
    const float* __restrict__ in, bf16* __restrict__ out, long n)
{
  const long i = (long)blockIdx.x * 256 + threadIdx.x;
  if (i < n) out[i] = (bf16)in[i];
}

__global__ __launch_bounds__(256) void layernorm_kernel(
    const float* __restrict__ x, const float* __restrict__ g,
    const float* __restrict__ b, bf16* __restrict__ out)
{
  __shared__ float red[256];
  const int s = blockIdx.x, tid = threadIdx.x;
  const float* xr = x + (long)s * D_DIM;
  float v[D_DIM / 256];
  float lsum = 0.f, lsq = 0.f;
#pragma unroll
  for (int i = 0; i < D_DIM / 256; ++i) {
    v[i] = xr[tid + i * 256];
    lsum += v[i]; lsq += v[i] * v[i];
  }
  red[tid] = lsum; __syncthreads();
  for (int off = 128; off > 0; off >>= 1) { if (tid < off) red[tid] += red[tid + off]; __syncthreads(); }
  const float mu = red[0] / (float)D_DIM;
  __syncthreads();
  red[tid] = lsq; __syncthreads();
  for (int off = 128; off > 0; off >>= 1) { if (tid < off) red[tid] += red[tid + off]; __syncthreads(); }
  const float var = red[0] / (float)D_DIM - mu * mu;
  const float rs = rsqrtf(var + LN_EPS);
#pragma unroll
  for (int i = 0; i < D_DIM / 256; ++i) {
    const int d = tid + i * 256;
    out[(long)s * D_DIM + d] = (bf16)((v[i] - mu) * rs * g[d] + b[d]);
  }
}

// causal softmax over row m (cols 0..m), writes bf16 probs, zeros beyond m
__global__ __launch_bounds__(256) void softmax_kernel(
    const float* __restrict__ sc, bf16* __restrict__ attn)
{
  __shared__ float red[256];
  const int m = blockIdx.x, tid = threadIdx.x;
  const float* row = sc + (long)m * S_LEN;
  float v[S_LEN / 256];
  float lmax = -3.4e38f;
#pragma unroll
  for (int i = 0; i < S_LEN / 256; ++i) {
    const int c = tid + i * 256;
    v[i] = (c <= m) ? row[c] : -3.4e38f;
    lmax = fmaxf(lmax, v[i]);
  }
  red[tid] = lmax; __syncthreads();
  for (int off = 128; off > 0; off >>= 1) { if (tid < off) red[tid] = fmaxf(red[tid], red[tid + off]); __syncthreads(); }
  const float mx = red[0];
  __syncthreads();
  float lsum = 0.f;
#pragma unroll
  for (int i = 0; i < S_LEN / 256; ++i) {
    const int c = tid + i * 256;
    const float e = (c <= m) ? __expf(v[i] - mx) : 0.f;
    v[i] = e; lsum += e;
  }
  red[tid] = lsum; __syncthreads();
  for (int off = 128; off > 0; off >>= 1) { if (tid < off) red[tid] += red[tid + off]; __syncthreads(); }
  const float inv = 1.f / red[0];
#pragma unroll
  for (int i = 0; i < S_LEN / 256; ++i) {
    const int c = tid + i * 256;
    attn[(long)m * S_LEN + c] = (bf16)(v[i] * inv);
  }
}

// v[S,D] bf16 -> vT[D,S] bf16 (so AV GEMM reuses the N-major layout)
__global__ __launch_bounds__(256) void transpose_bf16_kernel(
    const bf16* __restrict__ v, bf16* __restrict__ vt)
{
  __shared__ bf16 t[32][33];
  const int tx = threadIdx.x, ty = threadIdx.y;      // 32 x 8
  const long r0 = (long)blockIdx.y * 32;             // S rows
  const long c0 = (long)blockIdx.x * 32;             // D cols
#pragma unroll
  for (int k = 0; k < 4; ++k)
    t[ty + k * 8][tx] = v[(r0 + ty + k * 8) * D_DIM + c0 + tx];
  __syncthreads();
#pragma unroll
  for (int k = 0; k < 4; ++k)
    vt[(c0 + ty + k * 8) * S_LEN + r0 + tx] = (bf16)t[tx][ty + k * 8];
}

// ---------------- orchestration ----------------

extern "C" void kernel_launch(void* const* d_in, const int* in_sizes, int n_in,
                              void* d_out, int out_size, void* d_ws, size_t ws_size,
                              hipStream_t stream) {
  const int*   tokens  = (const int*)  d_in[0];
  const float* tok_emb = (const float*)d_in[1];
  const float* pos_emb = (const float*)d_in[2];
  const float* ln1_g   = (const float*)d_in[3];
  const float* ln1_b   = (const float*)d_in[4];
  const float* wq = (const float*)d_in[5];  const float* bq = (const float*)d_in[6];
  const float* wk = (const float*)d_in[7];  const float* bk = (const float*)d_in[8];
  const float* wv = (const float*)d_in[9];  const float* bv = (const float*)d_in[10];
  const float* wo = (const float*)d_in[11]; const float* bo = (const float*)d_in[12];
  const float* ln2_g = (const float*)d_in[13]; const float* ln2_b = (const float*)d_in[14];
  const float* w1 = (const float*)d_in[15]; const float* b1 = (const float*)d_in[16];
  const float* w2 = (const float*)d_in[17]; const float* b2 = (const float*)d_in[18];
  const float* head_w = (const float*)d_in[19];
  float* logits = (float*)d_out;

  char* ws = (char*)d_ws;
  size_t off = 0;
  auto alloc = [&](size_t bytes) -> void* {
    void* p = ws + off;
    off += (bytes + 255) & ~(size_t)255;
    return p;
  };

  const size_t SD = (size_t)S_LEN * D_DIM, SS = (size_t)S_LEN * S_LEN;
  const size_t SF = (size_t)S_LEN * F_DIM, DD = (size_t)D_DIM * D_DIM;
  const size_t FD = (size_t)F_DIM * D_DIM, VD = (size_t)V_DIM * D_DIM;

  float* x      = (float*)alloc(SD * 4);
  float* scores = (float*)alloc(SS * 4);
  float* x2     = (float*)alloc(SD * 4);
  bf16* h1    = (bf16*)alloc(SD * 2);
  bf16* qb    = (bf16*)alloc(SD * 2);
  bf16* kb    = (bf16*)alloc(SD * 2);
  bf16* vb    = (bf16*)alloc(SD * 2);
  bf16* vtb   = (bf16*)alloc(SD * 2);
  bf16* attnb = (bf16*)alloc(SS * 2);
  bf16* avb   = (bf16*)alloc(SD * 2);
  bf16* h2    = (bf16*)alloc(SD * 2);
  bf16* fb    = (bf16*)alloc(SF * 2);
  bf16* x3b   = (bf16*)alloc(SD * 2);
  bf16* wqb   = (bf16*)alloc(DD * 2);
  bf16* wkb   = (bf16*)alloc(DD * 2);
  bf16* wvb   = (bf16*)alloc(DD * 2);
  bf16* wob   = (bf16*)alloc(DD * 2);
  bf16* w1b   = (bf16*)alloc(FD * 2);
  bf16* w2b   = (bf16*)alloc(FD * 2);
  bf16* headb = (bf16*)alloc(VD * 2);

  const dim3 blk256(256);
  const dim3 gD(D_DIM / BN, S_LEN / BM);   // [S,D] outputs
  const dim3 gS(S_LEN / BN, S_LEN / BM);   // [S,S]
  const dim3 gF(F_DIM / BN, S_LEN / BM);   // [S,F]
  const dim3 gV(V_DIM / BN, S_LEN / BM);   // [S,V]  (32000 = 250*128)
  const float sc1 = 1.0f;
  const float scQK = 1.0f / sqrtf((float)D_DIM);

  // 0) weights -> bf16 (once; bf16 head fits L2)
  cvt_bf16_kernel<<<dim3((DD + 255) / 256), blk256, 0, stream>>>(wq, wqb, (long)DD);
  cvt_bf16_kernel<<<dim3((DD + 255) / 256), blk256, 0, stream>>>(wk, wkb, (long)DD);
  cvt_bf16_kernel<<<dim3((DD + 255) / 256), blk256, 0, stream>>>(wv, wvb, (long)DD);
  cvt_bf16_kernel<<<dim3((DD + 255) / 256), blk256, 0, stream>>>(wo, wob, (long)DD);
  cvt_bf16_kernel<<<dim3((FD + 255) / 256), blk256, 0, stream>>>(w1, w1b, (long)FD);
  cvt_bf16_kernel<<<dim3((FD + 255) / 256), blk256, 0, stream>>>(w2, w2b, (long)FD);
  cvt_bf16_kernel<<<dim3((VD + 255) / 256), blk256, 0, stream>>>(head_w, headb, (long)VD);

  // 1) x = tok_emb[tokens] + pos_emb
  embed_kernel<<<dim3(S_LEN), blk256, 0, stream>>>(tokens, tok_emb, pos_emb, x);

  // 2) h1 = LN1(x)
  layernorm_kernel<<<dim3(S_LEN), blk256, 0, stream>>>(x, ln1_g, ln1_b, h1);

  // 3) q/k/v (bf16 out, +bias)            <BIAS,RESID,RELU,OUTF,OUTB,CSKIP,CK>
  gemm_bf16_kernel<true,false,false,false,true,false,false><<<gD, blk256, 0, stream>>>(
      h1, wqb, nullptr, qb, bq, nullptr, D_DIM, D_DIM, sc1);
  gemm_bf16_kernel<true,false,false,false,true,false,false><<<gD, blk256, 0, stream>>>(
      h1, wkb, nullptr, kb, bk, nullptr, D_DIM, D_DIM, sc1);
  gemm_bf16_kernel<true,false,false,false,true,false,false><<<gD, blk256, 0, stream>>>(
      h1, wvb, nullptr, vb, bv, nullptr, D_DIM, D_DIM, sc1);

  // 4) vT for AV gemm
  transpose_bf16_kernel<<<dim3(D_DIM / 32, S_LEN / 32), dim3(32, 8), 0, stream>>>(vb, vtb);

  // 5) scores = (q @ k^T)/sqrt(D), skipping above-diagonal blocks
  gemm_bf16_kernel<false,false,false,true,false,true,false><<<gS, blk256, 0, stream>>>(
      qb, kb, scores, nullptr, nullptr, nullptr, S_LEN, D_DIM, scQK);

  // 6) causal softmax -> attn (bf16, zero-padded past diagonal)
  softmax_kernel<<<dim3(S_LEN), blk256, 0, stream>>>(scores, attnb);

  // 7) av = attn @ v (K clipped at diagonal), bf16 out
  gemm_bf16_kernel<false,false,false,false,true,false,true><<<gD, blk256, 0, stream>>>(
      attnb, vtb, nullptr, avb, nullptr, nullptr, D_DIM, S_LEN, sc1);

  // 8) x2 = x + av @ wo^T + bo
  gemm_bf16_kernel<true,true,false,true,false,false,false><<<gD, blk256, 0, stream>>>(
      avb, wob, x2, nullptr, bo, x, D_DIM, D_DIM, sc1);

  // 9) h2 = LN2(x2)
  layernorm_kernel<<<dim3(S_LEN), blk256, 0, stream>>>(x2, ln2_g, ln2_b, h2);

  // 10) f = relu(h2 @ w1^T + b1)
  gemm_bf16_kernel<true,false,true,false,true,false,false><<<gF, blk256, 0, stream>>>(
      h2, w1b, nullptr, fb, b1, nullptr, F_DIM, D_DIM, sc1);

  // 11) x3 = x2 + f @ w2^T + b2 (bf16 for head)
  gemm_bf16_kernel<true,true,false,false,true,false,false><<<gD, blk256, 0, stream>>>(
      fb, w2b, nullptr, x3b, b2, x2, D_DIM, F_DIM, sc1);

  // 12) logits = x3 @ head^T (f32 out)
  gemm_bf16_kernel<false,false,false,true,false,false,false><<<gV, blk256, 0, stream>>>(
      x3b, headb, logits, nullptr, nullptr, nullptr, V_DIM, D_DIM, sc1);

  (void)in_sizes; (void)n_in; (void)out_size; (void)ws_size;
}
